// LocalStructureLayer_55585466745465
// MI455X (gfx1250) — matrix operand order repaired and checked by reading, not compiled, not run on hardware
//
#include <hip/hip_runtime.h>

typedef __attribute__((ext_vector_type(16))) __bf16 v16bf;
typedef __attribute__((ext_vector_type(8)))  __bf16 v8bf;
typedef __attribute__((ext_vector_type(4)))  __bf16 v4bf;
typedef __attribute__((ext_vector_type(8)))  float  v8f;

#define C 128

// ---------------------------------------------------------------------------
// utility kernels
// ---------------------------------------------------------------------------
__global__ void fill_kernel(float* __restrict__ p, float val, int n) {
    int i = blockIdx.x * blockDim.x + threadIdx.x;
    if (i < n) p[i] = val;
}

// f32 -> bf16 convert, 4 elements per thread (n must be a multiple of 4)
__global__ void cvt_f32_bf16_kernel(const float* __restrict__ in,
                                    __bf16* __restrict__ out, int n) {
    int i = (blockIdx.x * blockDim.x + threadIdx.x) * 4;
    if (i >= n) return;
    const float4 f = *(const float4*)(in + i);
    v4bf o;
    o[0] = (__bf16)f.x; o[1] = (__bf16)f.y;
    o[2] = (__bf16)f.z; o[3] = (__bf16)f.w;
    *(v4bf*)(out + i) = o;
}

__device__ __forceinline__ void atomicMaxFloat(float* addr, float val) {
    // sign-aware float max via integer atomics (init value: large negative)
    if (val >= 0.0f) atomicMax((int*)addr, __float_as_int(val));
    else             atomicMin((unsigned int*)addr, __float_as_uint(val));
}

// ---------------------------------------------------------------------------
// WMMA GEMM: Out[M,128] = A @ W^T + bias, A/W pre-converted to bf16.
// block = 256 threads = 8 waves. Wave w owns the 16-column strip w*16..+15 and
// keeps all 4 K-step B fragments in registers; it sweeps 4 M-tiles (64 rows)
// per block so each B fragment is reused 4x. Inner loop: b128 loads + v_wmma.
// Epilogue: block-uniform scalar branch picks a branch-free store path for
// full tiles (only the final block can be partial).
// ---------------------------------------------------------------------------
__global__ void gemm_bf16_wmma(const __bf16* __restrict__ Ab,  // [M,128] bf16
                               const __bf16* __restrict__ Wb,  // [128,128] bf16 row-major
                               const float* __restrict__ bias, // [128]
                               float* __restrict__ Out, int M) {
    const int lane = threadIdx.x & 31;
    const int wave = threadIdx.x >> 5;
    const int rsel = lane >> 4;   // 0: lanes 0-15, 1: lanes 16-31
    const int l15  = lane & 15;

    // B fragment: lane holds column n of B = W^T (== row n of W);
    // half-slot h -> K = 16*rsel + h (contiguous k order, same as A)
    const int bcol = wave * 16 + l15;
    const int kbB  = 16 * rsel;
    const __bf16* __restrict__ Wrow = Wb + (size_t)bcol * C + kbB;
    const v16bf bf0 = *(const v16bf*)(Wrow + 0);
    const v16bf bf1 = *(const v16bf*)(Wrow + 32);
    const v16bf bf2 = *(const v16bf*)(Wrow + 64);
    const v16bf bf3 = *(const v16bf*)(Wrow + 96);
    const float bv  = bias[bcol];

    const int kbA = 16 * rsel;    // A half-slots 0..15 -> K=kbA+h (matches B layout)
    const int rowTile = blockIdx.x * 64;

#pragma unroll
    for (int t = 0; t < 4; ++t) {
        const int rowBase = rowTile + t * 16;
        if (rowBase >= M) break;                 // uniform branch, EXEC stays full
        int arow = rowBase + l15;
        if (arow >= M) arow = M - 1;             // clamp tail rows
        const __bf16* __restrict__ Arow = Ab + (size_t)arow * C + kbA;

        v8f acc = {};
#pragma unroll
        for (int kk = 0; kk < 4; ++kk) {
            const v8bf a0 = *(const v8bf*)(Arow + kk * 32);
            const v8bf a1 = *(const v8bf*)(Arow + kk * 32 + 8);
            v16bf a;
#pragma unroll
            for (int h = 0; h < 8; ++h) { a[h] = a0[h]; a[h + 8] = a1[h]; }
            const v16bf bf = (kk == 0) ? bf0 : (kk == 1) ? bf1 : (kk == 2) ? bf2 : bf3;
            acc = __builtin_amdgcn_wmma_f32_16x16x32_bf16(
                false, a, false, bf, (short)0, acc, false, false);
        }

        // D layout: VGPR r -> M = r + 8*rsel, N = l15
        float* __restrict__ Orow = Out + (size_t)(rowBase + 8 * rsel) * C + bcol;
        if (rowBase + 16 <= M) {
            // full tile: branch-free stores, constant row-stride offsets
#pragma unroll
            for (int r = 0; r < 8; ++r) Orow[(size_t)r * C] = acc[r] + bv;
        } else {
#pragma unroll
            for (int r = 0; r < 8; ++r) {
                if (rowBase + 8 * rsel + r < M) Orow[(size_t)r * C] = acc[r] + bv;
            }
        }
    }
}

// ---------------------------------------------------------------------------
// edge kernels (wave32 per edge; lane handles 4 channels)
// ---------------------------------------------------------------------------
__global__ void edge_alpha_kernel(const float* __restrict__ q,
                                  const float* __restrict__ k,
                                  const int* __restrict__ src,
                                  const int* __restrict__ dst,
                                  float* __restrict__ alpha,
                                  float* __restrict__ nmax, int E) {
    const int e = blockIdx.x * 8 + (threadIdx.x >> 5);
    if (e >= E) return;
    const int lane = threadIdx.x & 31;
    const int s = src[e], d = dst[e];
    const float4 qv = *(const float4*)(q + (size_t)d * C + lane * 4);
    const float4 kv = *(const float4*)(k + (size_t)s * C + lane * 4);
    float p = qv.x * kv.x + qv.y * kv.y + qv.z * kv.z + qv.w * kv.w;
#pragma unroll
    for (int off = 16; off; off >>= 1) p += __shfl_xor(p, off, 32);
    if (lane == 0) {
        const float al = p * 0.08838834764831845f; // 1/sqrt(128)
        alpha[e] = al;
        atomicMaxFloat(nmax + d, al);
    }
}

__global__ void edge_expsum_kernel(float* __restrict__ alpha,
                                   const float* __restrict__ nmax,
                                   const int* __restrict__ dst,
                                   float* __restrict__ nsum, int E) {
    const int e = blockIdx.x * blockDim.x + threadIdx.x;
    if (e >= E) return;
    const int d = dst[e];
    const float a = __expf(alpha[e] - nmax[d]);
    alpha[e] = a;
    atomicAdd(nsum + d, a);
}

__global__ void edge_scatter_kernel(const float* __restrict__ alpha,
                                    const float* __restrict__ nsum,
                                    const float* __restrict__ v,
                                    const int* __restrict__ src,
                                    const int* __restrict__ dst,
                                    float* __restrict__ attn, int E) {
    const int e = blockIdx.x * 8 + (threadIdx.x >> 5);
    if (e >= E) return;
    const int lane = threadIdx.x & 31;
    const int s = src[e], d = dst[e];
    const float w = alpha[e] / (nsum[d] + 1e-16f);
    const float4 vv = *(const float4*)(v + (size_t)s * C + lane * 4);
    float* o = attn + (size_t)d * C + lane * 4;
    atomicAdd(o + 0, w * vv.x);
    atomicAdd(o + 1, w * vv.y);
    atomicAdd(o + 2, w * vv.z);
    atomicAdd(o + 3, w * vv.w);
}

// ---------------------------------------------------------------------------
// LayerNorm (+ optional ReLU), wave per node; optionally also emits bf16 copy
// (staging input for the next layer's WMMA GEMMs)
// ---------------------------------------------------------------------------
__global__ void ln_kernel(const float* __restrict__ in,
                          const float* __restrict__ g,
                          const float* __restrict__ b,
                          float* __restrict__ out,
                          __bf16* __restrict__ outb,   // may be null
                          int N, int relu) {
    const int n = blockIdx.x * 8 + (threadIdx.x >> 5);
    if (n >= N) return;
    const int lane = threadIdx.x & 31;
    const float4 x = *(const float4*)(in + (size_t)n * C + lane * 4);
    float s = x.x + x.y + x.z + x.w;
#pragma unroll
    for (int off = 16; off; off >>= 1) s += __shfl_xor(s, off, 32);
    const float mu = s * (1.0f / C);
    const float d0 = x.x - mu, d1 = x.y - mu, d2 = x.z - mu, d3 = x.w - mu;
    float vs = d0 * d0 + d1 * d1 + d2 * d2 + d3 * d3;
#pragma unroll
    for (int off = 16; off; off >>= 1) vs += __shfl_xor(vs, off, 32);
    const float inv = rsqrtf(vs * (1.0f / C) + 1e-5f);
    const float4 gv = *(const float4*)(g + lane * 4);
    const float4 bv = *(const float4*)(b + lane * 4);
    float4 y;
    y.x = d0 * inv * gv.x + bv.x;
    y.y = d1 * inv * gv.y + bv.y;
    y.z = d2 * inv * gv.z + bv.z;
    y.w = d3 * inv * gv.w + bv.w;
    if (relu) {
        y.x = fmaxf(y.x, 0.0f); y.y = fmaxf(y.y, 0.0f);
        y.z = fmaxf(y.z, 0.0f); y.w = fmaxf(y.w, 0.0f);
    }
    *(float4*)(out + (size_t)n * C + lane * 4) = y;
    if (outb) {
        v4bf yb;
        yb[0] = (__bf16)y.x; yb[1] = (__bf16)y.y;
        yb[2] = (__bf16)y.z; yb[3] = (__bf16)y.w;
        *(v4bf*)(outb + (size_t)n * C + lane * 4) = yb;
    }
}

// ---------------------------------------------------------------------------
// orchestration
// ---------------------------------------------------------------------------
static inline size_t align256(size_t x) { return (x + 255) & ~(size_t)255; }

extern "C" void kernel_launch(void* const* d_in, const int* in_sizes, int n_in,
                              void* d_out, int out_size, void* d_ws, size_t ws_size,
                              hipStream_t stream) {
    const float* x  = (const float*)d_in[0];
    const int*   ei = (const int*)d_in[1];
    const int N = in_sizes[0] / C;
    const int E = in_sizes[1] / 2;
    const int* src = ei;
    const int* dst = ei + E;

    const size_t NB  = align256((size_t)N * C * sizeof(float));
    const size_t NBH = align256((size_t)N * C * sizeof(__bf16));
    char* w = (char*)d_ws;
    size_t off = 0;
    float* q     = (float*)(w + off); off += NB;
    float* kb_   = (float*)(w + off); off += NB;
    float* v     = (float*)(w + off); off += NB;
    float* attn  = (float*)(w + off); off += NB;
    float* hA    = (float*)(w + off); off += NB;
    float* hB    = (float*)(w + off); off += NB;
    float* alpha = (float*)(w + off); off += align256((size_t)E * sizeof(float));
    float* nmax  = (float*)(w + off); off += align256((size_t)N * sizeof(float));
    float* nsum  = (float*)(w + off); off += align256((size_t)N * sizeof(float));
    __bf16* hb   = (__bf16*)(w + off); off += NBH;
    __bf16* wb   = (__bf16*)(w + off); off += align256((size_t)12 * C * C * sizeof(__bf16));
    float* out   = (float*)d_out;

    // convert all 12 weight matrices to bf16 once per call
    for (int l = 0; l < 3; ++l)
        for (int j = 0; j < 4; ++j)
            cvt_f32_bf16_kernel<<<(C * C / 4 + 255) / 256, 256, 0, stream>>>(
                (const float*)d_in[2 + l * 10 + j], wb + (size_t)(l * 4 + j) * C * C, C * C);

    const int gemmBlocks     = (N + 63) / 64;
    const int edgeWaveBlocks = (E + 7) / 8;
    const int nodeWaveBlocks = (N + 7) / 8;
    const int cvtBlocks      = (N * C / 4 + 255) / 256;

    for (int l = 0; l < 3; ++l) {
        const int base = 2 + l * 10;
        const __bf16* Wq = wb + (size_t)(l * 4 + 0) * C * C;
        const __bf16* Wk = wb + (size_t)(l * 4 + 1) * C * C;
        const __bf16* Wv = wb + (size_t)(l * 4 + 2) * C * C;
        const __bf16* Wo = wb + (size_t)(l * 4 + 3) * C * C;
        const float* bq = (const float*)d_in[base + 4];
        const float* bk = (const float*)d_in[base + 5];
        const float* bv = (const float*)d_in[base + 6];
        const float* bo = (const float*)d_in[base + 7];
        const float* g  = (const float*)d_in[base + 8];
        const float* bb = (const float*)d_in[base + 9];

        float*  hout  = (l == 0) ? hA : (l == 1) ? hB : out;
        __bf16* houtb = (l < 2) ? hb : nullptr;   // bf16 staging for next layer
        const int relu = (l < 2) ? 1 : 0;

        fill_kernel<<<(N * C + 255) / 256, 256, 0, stream>>>(attn, 0.0f, N * C);
        fill_kernel<<<(N + 255) / 256, 256, 0, stream>>>(nmax, -3.0e38f, N);
        fill_kernel<<<(N + 255) / 256, 256, 0, stream>>>(nsum, 0.0f, N);

        if (l == 0)  // layer-1 activation input: x -> bf16 staging
            cvt_f32_bf16_kernel<<<cvtBlocks, 256, 0, stream>>>(x, hb, N * C);
        // layers 2/3: hb was written by the previous layer's ln_kernel

        gemm_bf16_wmma<<<gemmBlocks, 256, 0, stream>>>(hb, Wq, bq, q, N);
        gemm_bf16_wmma<<<gemmBlocks, 256, 0, stream>>>(hb, Wk, bk, kb_, N);
        gemm_bf16_wmma<<<gemmBlocks, 256, 0, stream>>>(hb, Wv, bv, v, N);

        edge_alpha_kernel<<<edgeWaveBlocks, 256, 0, stream>>>(q, kb_, src, dst, alpha, nmax, E);
        edge_expsum_kernel<<<(E + 255) / 256, 256, 0, stream>>>(alpha, nmax, dst, nsum, E);
        edge_scatter_kernel<<<edgeWaveBlocks, 256, 0, stream>>>(alpha, nsum, v, src, dst, attn, E);

        // attn (built by f32 atomics) -> bf16 staging, then output projection
        cvt_f32_bf16_kernel<<<cvtBlocks, 256, 0, stream>>>(attn, hb, N * C);
        gemm_bf16_wmma<<<gemmBlocks, 256, 0, stream>>>(hb, Wo, bo, q, N);  // q is dead: reuse
        ln_kernel<<<nodeWaveBlocks, 256, 0, stream>>>(q, g, bb, hout, houtb, N, relu);
    }
}
